// ScaleGraphBlock_17978733101637
// MI455X (gfx1250) — matrix operand order repaired and checked by reading, not compiled, not run on hardware
//
#include <hip/hip_runtime.h>
#include <math.h>

#define Bc   32
#define Tc   192
#define Nc   512
#define Vc   256
#define Hc   8
#define Ec   64      // Nc / Hc
#define DFFc 2048
#define NDc  40
#define CCx  16
#define SKc  32
#define FB   97      // T/2 + 1
#define LCAP 384
#define RMAX (Bc*LCAP)   // 12288 max padded rows
#define VT   (Vc*Tc)     // 49152

typedef __attribute__((ext_vector_type(8)))  unsigned v8u;
typedef __attribute__((ext_vector_type(16))) __bf16 v16bf;
typedef __attribute__((ext_vector_type(8)))  float v8f;
typedef unsigned short u16;

static __device__ __forceinline__ u16 f2bf(float f) {
  union { float f; unsigned u; } v; v.f = f;
  unsigned u = v.u;
  u += 0x7fffu + ((u >> 16) & 1u);     // round-to-nearest-even
  return (u16)(u >> 16);
}
static __device__ __forceinline__ float bf2f(u16 h) {
  union { unsigned u; float f; } v; v.u = ((unsigned)h) << 16;
  return v.f;
}
static __device__ __forceinline__ float gelu_f(float x) {
  return 0.5f * x * (1.0f + erff(x * 0.70710678118654752f));
}

// ---------------------------------------------------------------------------
// Batched GEMM on pre-staged bf16 operands via v_wmma_f32_16x16x32_bf16.
// Requirements (all launches satisfy): M_eff%16==0, N%32==0, K%32==0, unit
// K-stride for A and B, a_cb%32==0, strides multiples of 8 halves (16B).
//   val = cA*sum_k A*B + cR*R + bias[n]; optional gelu; store f32 C and/or
//   bf16 C2 (same layout).
// A addr (halves): m*a_rs + (k/a_cb)*a_cbs + (k%a_cb)   (+batch*a_bs)
// B addr (halves): k + n*b_ns                            (+batch*b_bs)
// Each wave: one 16-row x 32-col slab = two 16x16 WMMA tiles.
// Inner loop: 6x global_load_b128 + 2x v_wmma, zero conversion ALU.
// ---------------------------------------------------------------------------
struct GemmArgs {
  const u16* A; long long a_rs, a_cbs, a_bs; int a_cb;
  const u16* Bm; long long b_ns, b_bs;
  float* C; u16* C2; long long c_rs, c_cs, c_bs;
  const float* bias;
  const void* R; long long r_rs, r_cs, r_bs; int r_bf16;
  float cA, cR;
  int M, N, K;
  const int* dynM;   // optional device row bound (multiple of 16)
  int gelu;
};

__global__ __launch_bounds__(256) void gemm_wmma_kernel(GemmArgs g) {
  const int tn2 = (g.N >> 5);
  const int tm  = (g.M >> 4);
  const long long tiles = (long long)tm * tn2;
  const int wave = threadIdx.x >> 5;
  const long long tile = (long long)blockIdx.x * 8 + wave;
  if (tile >= tiles) return;                  // wave-uniform
  const int Me = g.dynM ? g.dynM[0] : g.M;
  const int ti = (int)(tile / tn2), tj = (int)(tile % tn2);
  const int row0 = ti << 4, col0 = tj << 5;
  if (row0 >= Me) return;                     // wave-uniform; rest in-bounds

  const int lane = threadIdx.x & 31;
  const int hi = lane >> 4;
  const int mi = lane & 15;

  const u16* Ab = g.A + (long long)blockIdx.y * g.a_bs;
  const u16* Bb = g.Bm + (long long)blockIdx.y * g.b_bs;

  // per-lane bases (halves)
  const u16* arow = Ab + (long long)(row0 + mi) * g.a_rs + (hi << 4);
  const u16* b0 = Bb + (long long)(col0 + mi) * g.b_ns + (hi << 4);
  const u16* b1 = Bb + (long long)(col0 + 16 + mi) * g.b_ns + (hi << 4);

  v8f acc0 = {0.f,0.f,0.f,0.f,0.f,0.f,0.f,0.f};
  v8f acc1 = {0.f,0.f,0.f,0.f,0.f,0.f,0.f,0.f};

  int blk = 0, rem = 0;
  for (int k0 = 0; k0 < g.K; k0 += 32) {
    const u16* ap = arow + (long long)blk * g.a_cbs + rem;
    uint4 ra0 = *(const uint4*)(ap);          // A halves {0..7}   (per-lane run 0)
    uint4 ra1 = *(const uint4*)(ap + 8);      // A halves {8..15}  (per-lane run 1)
    uint4 rb0 = *(const uint4*)(b0 + k0);
    uint4 rb1 = *(const uint4*)(b0 + k0 + 8);
    uint4 rc0 = *(const uint4*)(b1 + k0);
    uint4 rc1 = *(const uint4*)(b1 + k0 + 8);
    v8u ua; ua[0]=ra0.x; ua[1]=ra0.y; ua[2]=ra0.z; ua[3]=ra0.w;
            ua[4]=ra1.x; ua[5]=ra1.y; ua[6]=ra1.z; ua[7]=ra1.w;
    v8u ub; ub[0]=rb0.x; ub[1]=rb0.y; ub[2]=rb0.z; ub[3]=rb0.w;
            ub[4]=rb1.x; ub[5]=rb1.y; ub[6]=rb1.z; ub[7]=rb1.w;
    v8u uc; uc[0]=rc0.x; uc[1]=rc0.y; uc[2]=rc0.z; uc[3]=rc0.w;
            uc[4]=rc1.x; uc[5]=rc1.y; uc[6]=rc1.z; uc[7]=rc1.w;
    v16bf af  = __builtin_bit_cast(v16bf, ua);
    v16bf bf0 = __builtin_bit_cast(v16bf, ub);
    v16bf bf1 = __builtin_bit_cast(v16bf, uc);
    acc0 = __builtin_amdgcn_wmma_f32_16x16x32_bf16(false, af, false, bf0,
                                                   (short)0, acc0, false, false);
    acc1 = __builtin_amdgcn_wmma_f32_16x16x32_bf16(false, af, false, bf1,
                                                   (short)0, acc1, false, false);
    rem += 32;
    if (rem >= g.a_cb) { rem = 0; ++blk; }    // a_cb % 32 == 0, no div/mod
  }

  float* Cb = g.C ? (g.C + (long long)blockIdx.y * g.c_bs) : (float*)0;
  u16* C2b = g.C2 ? (g.C2 + (long long)blockIdx.y * g.c_bs) : (u16*)0;
  const char* Rb = g.R ? ((const char*)g.R) : (const char*)0;

#pragma unroll
  for (int r = 0; r < 8; ++r) {
    int m = row0 + r + (hi << 3);
    int n0 = col0 + mi;
    int n1 = col0 + 16 + mi;
    float v0 = g.cA * acc0[r];
    float v1 = g.cA * acc1[r];
    if (Rb) {
      long long o0 = (long long)blockIdx.y * g.r_bs + (long long)m * g.r_rs + (long long)n0 * g.r_cs;
      long long o1 = (long long)blockIdx.y * g.r_bs + (long long)m * g.r_rs + (long long)n1 * g.r_cs;
      if (g.r_bf16) { v0 += g.cR * bf2f(((const u16*)Rb)[o0]); v1 += g.cR * bf2f(((const u16*)Rb)[o1]); }
      else          { v0 += g.cR * ((const float*)Rb)[o0];      v1 += g.cR * ((const float*)Rb)[o1]; }
    }
    if (g.bias) { v0 += g.bias[n0]; v1 += g.bias[n1]; }
    if (g.gelu) { v0 = gelu_f(v0); v1 = gelu_f(v1); }
    long long c0 = (long long)m * g.c_rs + (long long)n0 * g.c_cs;
    long long c1 = (long long)m * g.c_rs + (long long)n1 * g.c_cs;
    if (Cb)  { Cb[c0] = v0; Cb[c1] = v1; }
    if (C2b) { C2b[c0] = f2bf(v0); C2b[c1] = f2bf(v1); }
  }
}

// -------------------------- f32 -> bf16 staging ----------------------------
__global__ void cvt_kernel(const float* src, u16* dst, long long n) {
  long long i = (long long)blockIdx.x * 256 + threadIdx.x;
  long long stride = (long long)gridDim.x * 256;
  for (; i < n; i += stride) dst[i] = f2bf(src[i]);
}

// ------------------------- DFT magnitude + top-k ---------------------------
__global__ void dft_kernel(const float* x, float* amp) {
  int f = blockIdx.x, b = blockIdx.y, tid = threadIdx.x;
  int n0 = tid, n1 = tid + 256;
  float re0=0.f, im0=0.f, re1=0.f, im1=0.f;
  for (int t = 0; t < Tc; ++t) {
    int ph = (f * t) % Tc;
    float sn, cs;
    __sincosf(-6.2831853071795864f * (float)ph / (float)Tc, &sn, &cs);
    const float* xr = x + ((long long)b * Tc + t) * Nc;
    float a0 = xr[n0], a1 = xr[n1];
    re0 += a0*cs; im0 += a0*sn; re1 += a1*cs; im1 += a1*sn;
  }
  float s = sqrtf(re0*re0 + im0*im0) + sqrtf(re1*re1 + im1*im1);
  __shared__ float red[256];
  red[tid] = s; __syncthreads();
  for (int o = 128; o > 0; o >>= 1) { if (tid < o) red[tid] += red[tid+o]; __syncthreads(); }
  if (tid == 0) amp[b*FB + f] = red[0] / (float)Nc;
}

__global__ void topk_kernel(const float* amp, int* meta, float* wgt) {
  if (threadIdx.x != 0) return;
  float freq[FB];
  for (int f = 0; f < FB; ++f) {
    float s = 0.f;
    for (int b = 0; b < Bc; ++b) s += amp[b*FB + f];
    freq[f] = s / (float)Bc;
  }
  freq[0] = 0.f;
  int top[2];
  for (int k = 0; k < 2; ++k) {
    int bi = 1; float bvv = -1.f;
    for (int f = 0; f < FB; ++f) if (freq[f] > bvv) { bvv = freq[f]; bi = f; }
    top[k] = bi; freq[bi] = -2.f;
  }
  for (int k = 0; k < 2; ++k) {
    int t = top[k] > 0 ? top[k] : 1;
    int scale = Tc / t; if (scale < 1) scale = 1;
    int len = ((Tc + scale - 1) / scale) * scale;
    meta[4*k+0] = scale;
    meta[4*k+1] = len;
    meta[4*k+2] = len / scale;     // segments per batch
    meta[4*k+3] = Bc * len;        // dynamic row count (multiple of 32)
  }
  for (int b = 0; b < Bc; ++b) {
    float a0 = amp[b*FB + top[0]], a1 = amp[b*FB + top[1]];
    float m = fmaxf(a0, a1);
    float e0 = expf(a0 - m), e1 = expf(a1 - m);
    float inv = 1.f / (e0 + e1);
    wgt[b*2+0] = e0*inv; wgt[b*2+1] = e1*inv;
  }
}

// --------------------- graph block helper kernels --------------------------
__global__ void adj_kernel(const float* nv1, const float* nv2, u16* a) {
  int v = blockIdx.x, w = threadIdx.x;
  float z = 0.f;
  for (int d = 0; d < NDc; ++d) z += nv1[v*NDc + d] * nv2[d*Vc + w];
  z = fmaxf(z, 0.f);
  __shared__ float red[256];
  red[w] = z; __syncthreads();
  for (int o = 128; o > 0; o >>= 1) { if (w < o) red[w] = fmaxf(red[w], red[w+o]); __syncthreads(); }
  float m = red[0]; __syncthreads();
  float p = expf(z - m);
  red[w] = p; __syncthreads();
  for (int o = 128; o > 0; o >>= 1) { if (w < o) red[w] += red[w+o]; __syncthreads(); }
  float s = red[0]; __syncthreads();
  float adjv = p / s + ((v == w) ? 1.f : 0.f);
  red[w] = adjv; __syncthreads();
  for (int o = 128; o > 0; o >>= 1) { if (w < o) red[w] += red[w+o]; __syncthreads(); }
  a[v*Vc + w] = f2bf(adjv / red[0]);
}

// h layout: [b][c][l][w]  (w contiguous) -> unit-stride K for mixprop GEMM
__global__ void scconv_kernel(const float* xin, const float* w, const float* bias, u16* bh0) {
  int t = blockIdx.x, b = blockIdx.y, tid = threadIdx.x;
  __shared__ float xs[Nc];
  const float* xr = xin + ((long long)b*Tc + t) * Nc;
  for (int n = tid; n < Nc; n += 256) xs[n] = xr[n];
  __syncthreads();
  int v = tid;  // Vc == 256 == blockDim
  for (int c = 0; c < CCx; ++c) {
    float acc = bias[c];
    const float* wc = w + c * 257;
    for (int i = 0; i < 257; ++i) acc += xs[v + i] * wc[i];
    bh0[(((long long)b*CCx + c)*Tc + t)*Vc + v] = f2bf(acc);
  }
}

__global__ void mlp_kernel(const u16* h0, const u16* h1, const u16* h2,
                           const float* mw, const float* mb, u16* ho2) {
  long long id = (long long)blockIdx.x * 256 + threadIdx.x;   // < B*V*T
  int b = (int)(id / (Vc*Tc));
  int rem = (int)(id % (Vc*Tc));
  int v = rem / Tc, l = rem % Tc;
  float hv[48];
  long long base = (long long)b*CCx*VT + (long long)l*Vc + v;  // h[b][c][l][v]
  for (int c = 0; c < CCx; ++c) {
    hv[c]      = bf2f(h0[base + (long long)c*VT]);
    hv[16 + c] = bf2f(h1[base + (long long)c*VT]);
    hv[32 + c] = bf2f(h2[base + (long long)c*VT]);
  }
  for (int s = 0; s < SKc; ++s) {
    float acc = mb[s];
    const float* ws = mw + s*48;
    for (int c = 0; c < 48; ++c) acc += ws[c] * hv[c];
    ho2[(((long long)b*SKc + s)*Vc + v)*Tc + l] = f2bf(gelu_f(acc));  // [b][s][v][l]
  }
}

// ------------------------------- layernorm ---------------------------------
__global__ void ln_kernel(const float* in, float* out, u16* out2,
                          const float* g, const float* b,
                          const int* dynRows, int do_gelu) {
  int row = blockIdx.x;
  if (dynRows && row >= dynRows[0]) return;
  int tid = threadIdx.x;
  const float* xr = in + (long long)row * Nc;
  float a0 = xr[tid], a1 = xr[tid+256];
  __shared__ float r1[256], r2[256];
  r1[tid] = a0 + a1; r2[tid] = a0*a0 + a1*a1;
  __syncthreads();
  for (int o = 128; o > 0; o >>= 1) {
    if (tid < o) { r1[tid] += r1[tid+o]; r2[tid] += r2[tid+o]; }
    __syncthreads();
  }
  float mean = r1[0] / (float)Nc;
  float var  = r2[0] / (float)Nc - mean*mean;
  float inv = rsqrtf(var + 1e-5f);
  float y0 = (a0 - mean)*inv*g[tid]     + b[tid];
  float y1 = (a1 - mean)*inv*g[tid+256] + b[tid+256];
  if (do_gelu) { y0 = gelu_f(y0); y1 = gelu_f(y1); }
  if (out) { float* orow = out + (long long)row * Nc; orow[tid] = y0; orow[tid+256] = y1; }
  if (out2) { u16* orow = out2 + (long long)row * Nc; orow[tid] = f2bf(y0); orow[tid+256] = f2bf(y1); }
}

// ------------------------- padding & attention core ------------------------
__global__ void pad_kernel(const float* xin, float* xp, u16* bxp, const int* mk) {
  int len = mk[1];
  int r = blockIdx.x;
  if (r >= Bc*len) return;
  int b = r / len, l = r % len;
  float* dst = xp + (long long)r * Nc;
  u16* bdst = bxp + (long long)r * Nc;
  if (l < Tc) {
    const float* src = xin + ((long long)b*Tc + l) * Nc;
    for (int n = threadIdx.x; n < Nc; n += 256) { float v = src[n]; dst[n] = v; bdst[n] = f2bf(v); }
  } else {
    for (int n = threadIdx.x; n < Nc; n += 256) { dst[n] = 0.f; bdst[n] = 0; }
  }
}

__global__ __launch_bounds__(256) void attn_kernel(const u16* qg, const u16* kg,
                                                   const u16* vg, u16* og,
                                                   const int* mk) {
  int L = mk[0], len = mk[1], nseg = mk[2];
  int s = blockIdx.x;
  if (s >= nseg) return;
  int b = blockIdx.y, h = blockIdx.z;
  long long base = (long long)b*len + (long long)s*L;
  __shared__ u16 kl[Tc*Ec];
  __shared__ u16 vl[Tc*Ec];
  __shared__ float pbuf[8][Tc];
  int tid = threadIdx.x;
  for (int idx = tid; idx < L*Ec; idx += 256) {
    int j = idx >> 6, e = idx & 63;
    kl[idx] = kg[(base + j)*Nc + h*Ec + e];
    vl[idx] = vg[(base + j)*Nc + h*Ec + e];
  }
  __syncthreads();
  int wave = tid >> 5, lane = tid & 31;
  float* pw = pbuf[wave];
  for (int qi = wave; qi < L; qi += 8) {      // one query row per wave per pass
    float qr[Ec];
    const u16* qrow = qg + (base + qi)*Nc + h*Ec;
#pragma unroll
    for (int e = 0; e < Ec; ++e) qr[e] = bf2f(qrow[e]);
    float m = -3.0e38f;
    for (int j = lane; j <= qi; j += 32) {    // causal: keys j <= qi
      float sc = 0.f;
      const u16* kj = kl + j*Ec;
#pragma unroll
      for (int e = 0; e < Ec; ++e) sc += qr[e] * bf2f(kj[e]);
      sc *= 0.125f;                            // 1/sqrt(64)
      pw[j] = sc;
      m = fmaxf(m, sc);
    }
    for (int o = 16; o > 0; o >>= 1) m = fmaxf(m, __shfl_xor(m, o, 32));
    float sum = 0.f;
    for (int j = lane; j <= qi; j += 32) {
      float p = expf(pw[j] - m);
      pw[j] = p;
      sum += p;
    }
    for (int o = 16; o > 0; o >>= 1) sum += __shfl_xor(sum, o, 32);
    float inv = 1.f / sum;
#pragma unroll
    for (int half = 0; half < 2; ++half) {
      int e = lane + half*32;
      float acc = 0.f;
      for (int j = 0; j <= qi; ++j) acc += pw[j] * bf2f(vl[j*Ec + e]);
      og[(base + qi)*Nc + h*Ec + e] = f2bf(acc * inv);
    }
  }
}

// ------------------------- weighted accumulation ---------------------------
__global__ void accum_kernel(float* out, const float* att, const float* xfin,
                             const float* wgt, const int* mk, int kidx, int addx) {
  int t = blockIdx.x, b = blockIdx.y;
  int len = mk[1];
  float w = wgt[b*2 + kidx];
  long long arow = ((long long)b*len + t) * Nc;
  long long orow = ((long long)b*Tc + t) * Nc;
  for (int n = threadIdx.x; n < Nc; n += 256) {
    float val = w * att[arow + n];
    if (kidx == 0) out[orow + n] = val;
    else {
      val += out[orow + n];
      if (addx) val += xfin[orow + n];
      out[orow + n] = val;
    }
  }
}

// ===========================================================================
extern "C" void kernel_launch(void* const* d_in, const int* in_sizes, int n_in,
                              void* d_out, int out_size, void* d_ws, size_t ws_size,
                              hipStream_t stream) {
  // dict order: x, then params flattened (sorted keys): att{W1,W2,Wk,Wo,Wq,Wv,
  // b1,b2,be1,be2,bk,bo,bq,bv,g1,g2}, gb[0..1]{b,ec_b,ec_w,g,lin_W,lin_b,mlp_b,
  // mlp_w,nv1,nv2,sc_b,sc_w}, norm_b, norm_g
  const float* x   = (const float*)d_in[0];
  const float* W1  = (const float*)d_in[1];
  const float* W2  = (const float*)d_in[2];
  const float* Wk  = (const float*)d_in[3];
  const float* Wo  = (const float*)d_in[4];
  const float* Wq  = (const float*)d_in[5];
  const float* Wv  = (const float*)d_in[6];
  const float* b1  = (const float*)d_in[7];
  const float* b2  = (const float*)d_in[8];
  const float* be1 = (const float*)d_in[9];
  const float* be2 = (const float*)d_in[10];
  const float* bk  = (const float*)d_in[11];
  const float* bo  = (const float*)d_in[12];
  const float* bq  = (const float*)d_in[13];
  const float* bv  = (const float*)d_in[14];
  const float* g1  = (const float*)d_in[15];
  const float* g2  = (const float*)d_in[16];
  const float* norm_b = (const float*)d_in[41];
  const float* norm_g = (const float*)d_in[42];

  // ---------------- workspace layout (256B-aligned chunks) ----------------
  char* base = (char*)d_ws;
  size_t off = 0;
  auto alloc = [&](size_t bytes) -> void* {
    off = (off + 255) & ~(size_t)255;
    void* p = base + off;
    off += bytes;
    return p;
  };
  int*   meta = (int*)  alloc(256);
  float* wgt  = (float*)alloc(256);
  float* amp  = (float*)alloc((size_t)Bc*FB*4);
  float* xc1  = (float*)alloc((size_t)Bc*Tc*Nc*4);
  float* xc2  = (float*)alloc((size_t)Bc*Tc*Nc*4);
  u16* bWq = (u16*)alloc((size_t)Nc*Nc*2);
  u16* bWk = (u16*)alloc((size_t)Nc*Nc*2);
  u16* bWv = (u16*)alloc((size_t)Nc*Nc*2);
  u16* bWo = (u16*)alloc((size_t)Nc*Nc*2);
  u16* bW1 = (u16*)alloc((size_t)DFFc*Nc*2);
  u16* bW2 = (u16*)alloc((size_t)Nc*DFFc*2);
  u16* bAdj = (u16*)alloc((size_t)Vc*Vc*2);
  u16* bEc  = (u16*)alloc((size_t)Tc*SKc*Tc*2);
  u16* bLin = (u16*)alloc((size_t)Nc*Vc*2);
  // f32 slab: graph uses pre; attention uses xp,t0,t1,t2 (phases sequential)
  size_t slabF_bytes = (size_t)4*RMAX*Nc*4;          // attn view is the larger
  char* slabF = (char*)alloc(slabF_bytes);
  float* pre = (float*)slabF;                        // [B][T][N]
  float* xp  = (float*)slabF;
  float* t0  = xp + (size_t)RMAX*Nc;
  float* t1  = t0 + (size_t)RMAX*Nc;
  float* t2  = t1 + (size_t)RMAX*Nc;
  // bf16 slab: graph {bH0,bH1,bH2,bHo2,bObuf} vs attn {bXp,bQ,bK,bV,bOb,bT1,bF1}
  size_t graphH = (size_t)(3*(size_t)Bc*CCx*VT + (size_t)Bc*SKc*VT + (size_t)Bc*Vc*Tc) * 2;
  size_t attnH  = (size_t)(6*(size_t)RMAX*Nc + (size_t)RMAX*DFFc) * 2;
  char* slabH = (char*)alloc(graphH > attnH ? graphH : attnH);
  u16* bH0   = (u16*)slabH;
  u16* bH1   = bH0 + (size_t)Bc*CCx*VT;
  u16* bH2   = bH1 + (size_t)Bc*CCx*VT;
  u16* bHo2  = bH2 + (size_t)Bc*CCx*VT;              // [b][s][v][l]
  u16* bObuf = bHo2 + (size_t)Bc*SKc*VT;             // [b][t][v]
  u16* bXp = (u16*)slabH;
  u16* bQ  = bXp + (size_t)RMAX*Nc;
  u16* bK  = bQ + (size_t)RMAX*Nc;
  u16* bV  = bK + (size_t)RMAX*Nc;
  u16* bOb = bV + (size_t)RMAX*Nc;
  u16* bT1 = bOb + (size_t)RMAX*Nc;
  u16* bF1 = bT1 + (size_t)RMAX*Nc;
  if (ws_size < off) return;

  float* out = (float*)d_out;

  auto gemm = [&](const u16* A, long long a_rs, int a_cb, long long a_cbs, long long a_bs,
                  const u16* Bm, long long b_ns, long long b_bs,
                  float* C, u16* C2, long long c_rs, long long c_cs, long long c_bs,
                  const float* bias, const void* R, long long r_rs, long long r_cs,
                  long long r_bs, int r_bf16,
                  float cA, float cR, int M, int N, int K, const int* dynM, int gl, int batch) {
    GemmArgs ga;
    ga.A = A; ga.a_rs = a_rs; ga.a_cb = a_cb; ga.a_cbs = a_cbs; ga.a_bs = a_bs;
    ga.Bm = Bm; ga.b_ns = b_ns; ga.b_bs = b_bs;
    ga.C = C; ga.C2 = C2; ga.c_rs = c_rs; ga.c_cs = c_cs; ga.c_bs = c_bs;
    ga.bias = bias; ga.R = R; ga.r_rs = r_rs; ga.r_cs = r_cs; ga.r_bs = r_bs; ga.r_bf16 = r_bf16;
    ga.cA = cA; ga.cR = cR; ga.M = M; ga.N = N; ga.K = K; ga.dynM = dynM; ga.gelu = gl;
    long long tiles = (long long)(M / 16) * (N / 32);
    int blocks = (int)((tiles + 7) / 8);
    gemm_wmma_kernel<<<dim3(blocks, batch), dim3(256), 0, stream>>>(ga);
  };

  // stage attention weights to bf16 once
  cvt_kernel<<<512, 256, 0, stream>>>(Wq, bWq, (long long)Nc*Nc);
  cvt_kernel<<<512, 256, 0, stream>>>(Wk, bWk, (long long)Nc*Nc);
  cvt_kernel<<<512, 256, 0, stream>>>(Wv, bWv, (long long)Nc*Nc);
  cvt_kernel<<<512, 256, 0, stream>>>(Wo, bWo, (long long)Nc*Nc);
  cvt_kernel<<<2048, 256, 0, stream>>>(W1, bW1, (long long)DFFc*Nc);
  cvt_kernel<<<2048, 256, 0, stream>>>(W2, bW2, (long long)Nc*DFFc);

  dft_kernel<<<dim3(FB, Bc), 256, 0, stream>>>(x, amp);
  topk_kernel<<<1, 64, 0, stream>>>(amp, meta, wgt);

  for (int k = 0; k < 2; ++k) {
    int gi = 17 + 12*k;
    const float* gb_b  = (const float*)d_in[gi+0];
    const float* ec_b  = (const float*)d_in[gi+1];
    const float* ec_w  = (const float*)d_in[gi+2];
    const float* gb_g  = (const float*)d_in[gi+3];
    const float* lin_W = (const float*)d_in[gi+4];
    const float* lin_b = (const float*)d_in[gi+5];
    const float* mlp_b = (const float*)d_in[gi+6];
    const float* mlp_w = (const float*)d_in[gi+7];
    const float* nv1   = (const float*)d_in[gi+8];
    const float* nv2   = (const float*)d_in[gi+9];
    const float* sc_b  = (const float*)d_in[gi+10];
    const float* sc_w  = (const float*)d_in[gi+11];
    const float* xin = (k == 0) ? x : xc1;
    float* xout = (k == 0) ? xc1 : xc2;
    const int* mk = meta + 4*k;
    const int* dynR = mk + 3;

    // ---------------- graph block ----------------
    cvt_kernel<<<2048, 256, 0, stream>>>(ec_w, bEc, (long long)Tc*SKc*Tc);
    cvt_kernel<<<512, 256, 0, stream>>>(lin_W, bLin, (long long)Nc*Vc);
    adj_kernel<<<Vc, Vc, 0, stream>>>(nv1, nv2, bAdj);
    scconv_kernel<<<dim3(Tc, Bc), 256, 0, stream>>>(xin, sc_w, sc_b, bH0);
    // mixprop: h_{s+1}[v,l] = 0.05*h0 + 0.95*sum_w a[v,w]*h_s[l,w]  (batch b*c)
    gemm(bAdj, Vc, Vc, 0, 0,
         bH0, Vc, VT,
         nullptr, bH1, 1, Vc, VT,
         nullptr, bH0, 1, Vc, VT, 1,
         0.95f, 0.05f, Vc, Tc, Vc, nullptr, 0, Bc*CCx);
    gemm(bAdj, Vc, Vc, 0, 0,
         bH1, Vc, VT,
         nullptr, bH2, 1, Vc, VT,
         nullptr, bH0, 1, Vc, VT, 1,
         0.95f, 0.05f, Vc, Tc, Vc, nullptr, 0, Bc*CCx);
    mlp_kernel<<<(Bc*Vc*Tc)/256, 256, 0, stream>>>(bH0, bH1, bH2, mlp_w, mlp_b, bHo2);
    // ec: per b, D[v,t] = sum_{q=(s,l)} ho2[b,s,v,l]*ec_w[t,q] + ec_b[t]
    gemm(bHo2, Tc, Tc, VT, (long long)SKc*VT,
         bEc, (long long)SKc*Tc, 0,
         nullptr, bObuf, 1, Vc, VT,            // store [t][v] bf16
         ec_b, nullptr, 0, 0, 0, 0, 1.f, 0.f, Vc, Tc, SKc*Tc, nullptr, 0, Bc);
    // lin + residual: pre[b,t,n] = xin + obuf[b,t,:] @ lin_W^T + lin_b
    gemm(bObuf, Vc, Vc, 0, VT,
         bLin, Vc, 0,
         pre, nullptr, Nc, 1, (long long)Tc*Nc,
         lin_b, xin, Nc, 1, (long long)Tc*Nc, 0,
         1.f, 1.f, Tc, Nc, Vc, nullptr, 0, Bc);
    ln_kernel<<<Bc*Tc, 256, 0, stream>>>(pre, xout, nullptr, gb_g, gb_b, nullptr, 0);

    // ---------------- attention over dynamic segments ----------------
    pad_kernel<<<RMAX, 256, 0, stream>>>(xout, xp, bXp, mk);
    gemm(bXp, Nc, Nc, 0, 0, bWq, Nc, 0, nullptr, bQ, Nc, 1, 0,
         bq, nullptr, 0, 0, 0, 0, 1.f, 0.f, RMAX, Nc, Nc, dynR, 0, 1);
    gemm(bXp, Nc, Nc, 0, 0, bWk, Nc, 0, nullptr, bK, Nc, 1, 0,
         bk, nullptr, 0, 0, 0, 0, 1.f, 0.f, RMAX, Nc, Nc, dynR, 0, 1);
    gemm(bXp, Nc, Nc, 0, 0, bWv, Nc, 0, nullptr, bV, Nc, 1, 0,
         bv, nullptr, 0, 0, 0, 0, 1.f, 0.f, RMAX, Nc, Nc, dynR, 0, 1);
    attn_kernel<<<dim3(96, Bc, Hc), 256, 0, stream>>>(bQ, bK, bV, bOb, mk);
    gemm(bOb, Nc, Nc, 0, 0, bWo, Nc, 0, t0, nullptr, Nc, 1, 0,
         bo, xp, Nc, 1, 0, 0, 1.f, 1.f, RMAX, Nc, Nc, dynR, 0, 1);
    ln_kernel<<<RMAX, 256, 0, stream>>>(t0, t1, bT1, g1, be1, dynR, 0);
    gemm(bT1, Nc, Nc, 0, 0, bW1, Nc, 0, nullptr, bF1, DFFc, 1, 0,
         b1, nullptr, 0, 0, 0, 0, 1.f, 0.f, RMAX, DFFc, Nc, dynR, 1, 1);
    gemm(bF1, DFFc, DFFc, 0, 0, bW2, DFFc, 0, t0, nullptr, Nc, 1, 0,
         b2, t1, Nc, 1, 0, 0, 1.f, 1.f, RMAX, Nc, DFFc, dynR, 0, 1);
    ln_kernel<<<RMAX, 256, 0, stream>>>(t0, t2, nullptr, g2, be2, dynR, 0);
    ln_kernel<<<RMAX, 256, 0, stream>>>(t2, t0, nullptr, norm_g, norm_b, dynR, 1);
    accum_kernel<<<dim3(Tc, Bc), 256, 0, stream>>>(out, t0, xc2, wgt, mk, k, (k == 1) ? 1 : 0);
  }
}
